// GraphConvolutionalLayer_21672404976273
// MI455X (gfx1250) — matrix operand order repaired and checked by reading, not compiled, not run on hardware
//
#include <hip/hip_runtime.h>
#include <hip/hip_bf16.h>

typedef __attribute__((ext_vector_type(16))) _Float16 v16h;
typedef __attribute__((ext_vector_type(8)))  float    v8f;

#define IN_F  256
#define OUT_F 128

// -------------------------------------------------------------------------
// Kernel 1: convert w [IN_F][OUT_F] fp32  ->  wT [OUT_F][IN_F] f16
// (K contiguous per output column, so B-fragments are contiguous 16-half runs)
// -------------------------------------------------------------------------
__global__ void gcn_convert_w(const float* __restrict__ w, _Float16* __restrict__ wT) {
    int tid = blockIdx.x * blockDim.x + threadIdx.x;     // IN_F*OUT_F threads
    int k = tid & (IN_F - 1);
    int n = tid >> 8;                                    // IN_F == 256
    if (n < OUT_F)
        wT[(size_t)n * IN_F + k] = (_Float16)w[(size_t)k * OUT_F + n];
}

// -------------------------------------------------------------------------
// Kernel 2: h = x @ w via v_wmma_f32_16x16x32_f16.
// One wave per 16-row M tile, full N=128 (8 accumulator tiles, 64 VGPRs).
// A fragment layout (16-bit A, 16x32):
//   lanes 0-15 : M=lane,    halves 0..15 = K 0..15
//   lanes 16-31: M=lane-16, halves 0..15 = K 16..31
// B fragment layout (16-bit B, 32x16):
//   lanes 0-15 : N=lane,    halves = K 0..15 ; lanes 16-31: K 16..31
// h stored as f16 (halves the SpMM gather traffic; L2-resident at 25.6 MB).
// -------------------------------------------------------------------------
__global__ void __launch_bounds__(256)
gcn_gemm_wmma(const float* __restrict__ x, const _Float16* __restrict__ wT,
              _Float16* __restrict__ hout, int ntiles) {
    int wave = blockIdx.x * (blockDim.x >> 5) + (threadIdx.x >> 5);
    if (wave >= ntiles) return;                 // wave-uniform: EXEC stays all-1s
    int lane = threadIdx.x & 31;
    int l15  = lane & 15;
    int hi   = lane >> 4;                       // 0 or 1 (half-wave)
    int mbase = wave * 16;

    const float* xrow = x + (size_t)(mbase + l15) * IN_F;

    v8f zero = {};
    v8f acc[8];
#pragma unroll
    for (int n = 0; n < 8; ++n) acc[n] = zero;

#pragma unroll
    for (int kc = 0; kc < IN_F / 32; ++kc) {
        int base0 = kc * 32 + hi * 16;          // K start for halves 0..15
        // 8 + 8 fp32 loads (4x float4, 32B aligned), convert to f16 A fragment
        const float4* p0 = (const float4*)(xrow + base0);
        float4 f0 = p0[0], f1 = p0[1];
        const float4* p1 = (const float4*)(xrow + base0 + 8);
        float4 f2 = p1[0], f3 = p1[1];

        if (kc + 1 < IN_F / 32)                 // near-cache prefetch of next A chunk
            __builtin_prefetch(xrow + base0 + 32, 0, 3);

        v16h a;
        a[0]  = (_Float16)f0.x; a[1]  = (_Float16)f0.y;
        a[2]  = (_Float16)f0.z; a[3]  = (_Float16)f0.w;
        a[4]  = (_Float16)f1.x; a[5]  = (_Float16)f1.y;
        a[6]  = (_Float16)f1.z; a[7]  = (_Float16)f1.w;
        a[8]  = (_Float16)f2.x; a[9]  = (_Float16)f2.y;
        a[10] = (_Float16)f2.z; a[11] = (_Float16)f2.w;
        a[12] = (_Float16)f3.x; a[13] = (_Float16)f3.y;
        a[14] = (_Float16)f3.z; a[15] = (_Float16)f3.w;

        int kb = kc * 32 + hi * 16;             // B fragment K base (32B aligned)
#pragma unroll
        for (int n = 0; n < 8; ++n) {
            const v16h* bp = (const v16h*)(wT + (size_t)(n * 16 + l15) * IN_F + kb);
            v16h b = *bp;
            acc[n] = __builtin_amdgcn_wmma_f32_16x16x32_f16(
                /*neg_a=*/false, a, /*neg_b=*/false, b,
                /*c_mod=*/(short)0, acc[n], /*reuse_a=*/false, /*reuse_b=*/false);
        }
    }

    // C/D layout: VGPR j, lanes 0-15 -> M = mbase+j, N = l15 (per 16-col tile);
    // lanes 16-31 -> M = mbase+8+j. Store as f16.
#pragma unroll
    for (int n = 0; n < 8; ++n) {
        int col = n * 16 + l15;
#pragma unroll
        for (int j = 0; j < 8; ++j) {
            int row = mbase + j + hi * 8;
            hout[(size_t)row * OUT_F + col] = (_Float16)acc[n][j];
        }
    }
}

// Scalar fallback for remainder rows (nnodes % 16 — zero for this shape).
__global__ void gcn_gemm_tail(const float* __restrict__ x, const float* __restrict__ w,
                              _Float16* __restrict__ hout, int nnodes, int start) {
    int idx = blockIdx.x * blockDim.x + threadIdx.x;
    int row = start + idx / OUT_F;
    int col = idx % OUT_F;
    if (row >= nnodes) return;
    float s = 0.f;
    for (int k = 0; k < IN_F; ++k)
        s += x[(size_t)row * IN_F + k] * w[(size_t)k * OUT_F + col];
    hout[(size_t)row * OUT_F + col] = (_Float16)s;
}

// -------------------------------------------------------------------------
// Kernel 3: out = bias (broadcast init before scatter-accumulate)
// -------------------------------------------------------------------------
__global__ void gcn_init_out(const float* __restrict__ bias, float* __restrict__ out,
                             long long total) {
    long long t = blockIdx.x * (long long)blockDim.x + threadIdx.x;
    if (t < total) out[t] = bias[t & (OUT_F - 1)];
}

// -------------------------------------------------------------------------
// Kernel 4: edge scatter. ONE WAVE PER EDGE: 32 lanes x 4 features.
// Edge index is wave-uniform; readfirstlane forces it into an SGPR so the
// row/col/val loads become scalar (s_load) instead of redundant VMEM.
// Lane handles features lane + 32*i -> each atomic instruction covers a
// contiguous 128B span across the wave; gathers are contiguous 64B spans.
// -------------------------------------------------------------------------
__global__ void __launch_bounds__(256)
gcn_scatter(const _Float16* __restrict__ h, const int* __restrict__ erow,
            const int* __restrict__ ecol, const float* __restrict__ ev,
            float* __restrict__ out, int nedges) {
    int e = blockIdx.x * (blockDim.x >> 5) + (threadIdx.x >> 5);
    if (e >= nedges) return;                    // wave-uniform branch
    int lane = threadIdx.x & 31;

    int eu = __builtin_amdgcn_readfirstlane(e); // SGPR edge index -> scalar loads
    int   r = erow[eu];
    int   c = ecol[eu];
    float v = ev[eu];

    const _Float16* hrow = h   + (size_t)c * OUT_F;
    float*          orow = out + (size_t)r * OUT_F;

#pragma unroll
    for (int i = 0; i < OUT_F / 32; ++i) {      // 4 features per lane
        int f = lane + 32 * i;
        float hv = (float)hrow[f];
        atomicAdd(orow + f, v * hv);
    }
}

// -------------------------------------------------------------------------
// Launcher. Inputs: x, edge_row, edge_col, edge_val, w, bias.
// Workspace: [h f16: nnodes*OUT_F*2 bytes][wT f16: OUT_F*IN_F*2 bytes]
// -------------------------------------------------------------------------
extern "C" void kernel_launch(void* const* d_in, const int* in_sizes, int n_in,
                              void* d_out, int out_size, void* d_ws, size_t ws_size,
                              hipStream_t stream) {
    const float* x     = (const float*)d_in[0];
    const int*   erow  = (const int*)  d_in[1];
    const int*   ecol  = (const int*)  d_in[2];
    const float* ev    = (const float*)d_in[3];
    const float* w     = (const float*)d_in[4];
    const float* bias  = (const float*)d_in[5];
    float*       out   = (float*)d_out;

    int nnodes = in_sizes[0] / IN_F;
    int nedges = in_sizes[1];

    _Float16* h  = (_Float16*)d_ws;
    _Float16* wT = (_Float16*)((char*)d_ws + (size_t)nnodes * OUT_F * sizeof(_Float16));

    // 1) w -> wT (f16, transposed)
    gcn_convert_w<<<(IN_F * OUT_F + 255) / 256, 256, 0, stream>>>(w, wT);

    // 2) h = x @ w via WMMA (one wave per 16-row tile, 8 waves/block)
    int ntiles = nnodes / 16;
    if (ntiles > 0) {
        int blocks = (ntiles + 7) / 8;
        gcn_gemm_wmma<<<blocks, 256, 0, stream>>>(x, wT, h, ntiles);
    }
    int rem = nnodes - ntiles * 16;
    if (rem > 0) {
        int tail_threads = rem * OUT_F;
        gcn_gemm_tail<<<(tail_threads + 255) / 256, 256, 0, stream>>>(
            x, w, h, nnodes, ntiles * 16);
    }

    // 3) out = bias
    long long otot = (long long)nnodes * OUT_F;
    gcn_init_out<<<(unsigned)((otot + 255) / 256), 256, 0, stream>>>(bias, out, otot);

    // 4) scatter-accumulate: one wave per edge
    int sblocks = (nedges + 7) / 8;
    gcn_scatter<<<sblocks, 256, 0, stream>>>(h, erow, ecol, ev, out, nedges);
}
